// EnhancedGCNEncoder_67156108640277
// MI455X (gfx1250) — matrix operand order repaired and checked by reading, not compiled, not run on hardware
//
#include <hip/hip_runtime.h>
#include <math.h>

// ---------------------------------------------------------------------------
// Types for CDNA5 WMMA (wave32, bf16 -> f32 accumulate)
// ---------------------------------------------------------------------------
typedef __attribute__((ext_vector_type(16))) __bf16 v16bf;
typedef __attribute__((ext_vector_type(8)))  float  v8f;

union ABFrag { uint4 u[2]; v16bf v; };   // 32 bytes: 16 bf16 per lane

__device__ __forceinline__ unsigned short f2bf(float f) {
  union { float f; unsigned u; } c; c.f = f;
  unsigned r = c.u + 0x7FFFu + ((c.u >> 16) & 1u);   // round-to-nearest-even
  return (unsigned short)(r >> 16);
}

// Problem constants
#define NN   100000
#define NE   1600000
#define KDIM 256           // K is 256 for both layers
#define MT   (NN / 16)     // 6250 m-tiles (exact)
#define KB   (KDIM / 32)   // 8 WMMA K-steps

// ---------------------------------------------------------------------------
// Degree / normalization
// ---------------------------------------------------------------------------
__global__ void deg_init_k(float* __restrict__ deg) {
  int i = blockIdx.x * blockDim.x + threadIdx.x;
  if (i < NN) deg[i] = 1.0f;                       // self-loop weight
}

__global__ void deg_scatter_k(const long long* __restrict__ ei,
                              const float* __restrict__ ew,
                              float* __restrict__ deg) {
  int e = blockIdx.x * blockDim.x + threadIdx.x;
  if (e < NE) {
    long long d = ei[NE + e];
    unsafeAtomicAdd(&deg[d], ew[e]);
  }
}

__global__ void dinv_k(float* __restrict__ deg) {
  int i = blockIdx.x * blockDim.x + threadIdx.x;
  if (i < NN) {
    float d = deg[i];
    deg[i] = (d > 0.0f) ? rsqrtf(d) : 0.0f;        // in-place deg -> dinv
  }
}

// ---------------------------------------------------------------------------
// Pack activations (f32, row-major [rows,256]) into per-lane A-fragment order.
// A 16x32 bf16 layout: lane L(0-15): M=L, K=0..15; lane 16-31:
// M=L-16, K=16..31; element j -> K = kbase + j (contiguous, matches B).
// Packed blob per (mtile, kstep): 32 lanes * 16 bf16 = 1024B, lane-contiguous.
// Optionally applies exact-erf GELU (layer-1 epilogue fused here).
// ---------------------------------------------------------------------------
__global__ void pack_a_k(const float* __restrict__ src,
                         unsigned short* __restrict__ dst,
                         int apply_gelu) {
  int p = blockIdx.x * blockDim.x + threadIdx.x;   // total = MT*4096 = 25.6M
  if (p >= MT * 4096) return;
  int j    = p & 15;
  int lane = (p >> 4) & 31;
  int kb   = (p >> 9) & 7;
  int mt   = p >> 12;
  int m    = mt * 16 + (lane & 15);
  int koff = (lane >> 4) << 3;
  int kl   = ((lane >> 4) << 4) + j;
  float v  = src[m * KDIM + (kb * 32 + kl)];
  if (apply_gelu) v = 0.5f * v * (1.0f + erff(v * 0.70710678118654752f));
  dst[p] = f2bf(v);
}

// ---------------------------------------------------------------------------
// Pack weights (f32, row-major [K,N]) into per-lane B-fragment order.
// B 32x16 bf16 layout: lane L(0-15): N=L, K=0..15 ; lane 16-31: N=L-16,
// K=16..31 ; element j -> K = kbase + j (contiguous).
// ---------------------------------------------------------------------------
__global__ void pack_b_k(const float* __restrict__ W,
                         unsigned short* __restrict__ dst, int N) {
  int p = blockIdx.x * blockDim.x + threadIdx.x;   // total = K*N
  if (p >= KDIM * N) return;
  int j    = p & 15;
  int lane = (p >> 4) & 31;
  int kb   = (p >> 9) & 7;
  int nb   = p >> 12;
  int n    = lane & 15;
  int k    = kb * 32 + ((lane >> 4) << 4) + j;
  dst[p] = f2bf(W[k * N + nb * 16 + n]);
}

// ---------------------------------------------------------------------------
// BF16 WMMA GEMM: H[NN,NCOLS] = Apk(bf16) @ Bpk(bf16), f32 accumulate.
// One wave per 16-row m-tile; blockIdx.y selects a 128-column n-chunk so each
// wave owns NT=8 n-subtiles -> 64 accumulator VGPRs, no scratch spills.
// Fragment loads are two coalesced b128 per lane. Wave-uniform guard keeps
// EXEC all-ones as required by WMMA.
// ---------------------------------------------------------------------------
template <int NCOLS>
__global__ void gemm_wmma_k(const uint4* __restrict__ Apk,
                            const uint4* __restrict__ Bpk,
                            float* __restrict__ H) {
  constexpr int NT = 8;                            // 128 columns per wave
  int wave = threadIdx.x >> 5;
  int lane = threadIdx.x & 31;
  int mt = blockIdx.x * 8 + wave;
  int nt0 = blockIdx.y * NT;                       // first n-subtile
  if (mt >= MT) return;                            // uniform per wave

  v8f acc[NT];
#pragma unroll
  for (int i = 0; i < NT; ++i) acc[i] = (v8f){0,0,0,0,0,0,0,0};

#pragma unroll
  for (int kb = 0; kb < KB; ++kb) {
    ABFrag fa;
    int ab = ((mt * KB + kb) * 32 + lane) * 2;
    fa.u[0] = Apk[ab];
    fa.u[1] = Apk[ab + 1];
#pragma unroll
    for (int nt = 0; nt < NT; ++nt) {
      ABFrag fb;
      int bb = (((nt0 + nt) * KB + kb) * 32 + lane) * 2;
      fb.u[0] = Bpk[bb];
      fb.u[1] = Bpk[bb + 1];
      acc[nt] = __builtin_amdgcn_wmma_f32_16x16x32_bf16(
          false, fa.v, false, fb.v, (short)0, acc[nt], false, false);
    }
  }

  // C/D layout: element r -> row m0+r, col = (nt0+nt)*16 + (lane&15);
  // lanes 16-31 cover rows +8.
  int m0 = mt * 16 + ((lane >> 4) << 3);
  int n  = lane & 15;
#pragma unroll
  for (int nt = 0; nt < NT; ++nt) {
    int c = (nt0 + nt) * 16 + n;
#pragma unroll
    for (int r = 0; r < 8; ++r) H[(m0 + r) * NCOLS + c] = acc[nt][r];
  }
}

// ---------------------------------------------------------------------------
// Aggregation init: out[i,c] = bias[c] + dinv[i]^2 * h[i,c]
// (self-loop norm = dinv^2, bias folded in — added exactly once per node)
// ---------------------------------------------------------------------------
template <int C>
__global__ void agg_init_k(const float* __restrict__ h,
                           const float* __restrict__ dinv,
                           const float* __restrict__ bias,
                           float* __restrict__ out) {
  int idx = blockIdx.x * blockDim.x + threadIdx.x;
  if (idx >= NN * C) return;
  int row = idx / C;
  int c   = idx & (C - 1);
  float di = dinv[row];
  out[idx] = bias[c] + di * di * h[idx];
}

// ---------------------------------------------------------------------------
// Edge scatter: one wave per edge; out[dst,c] += dinv[src]*ew*dinv[dst]*h[src,c]
// h[src] rows are L2-resident (h1 = 102.4MB, h2 = 51.2MB < 192MB L2).
// Native f32 atomics (fire-and-forget) to d_out/agg.
// ---------------------------------------------------------------------------
template <int C>
__global__ void edge_scatter_k(const long long* __restrict__ ei,
                               const float* __restrict__ ew,
                               const float* __restrict__ dinv,
                               const float* __restrict__ h,
                               float* __restrict__ out) {
  int e    = (blockIdx.x * blockDim.x + threadIdx.x) >> 5;  // wave per edge
  int lane = threadIdx.x & 31;
  if (e >= NE) return;                                      // uniform per wave
  long long s = ei[e];
  long long d = ei[NE + e];
  float nrm = dinv[s] * ew[e] * dinv[d];
  const float4* hs = (const float4*)(h + (size_t)s * C);
  float* od = out + (size_t)d * C;
#pragma unroll
  for (int v = lane; v < (C >> 2); v += 32) {
    float4 val = hs[v];
    unsafeAtomicAdd(od + v * 4 + 0, nrm * val.x);
    unsafeAtomicAdd(od + v * 4 + 1, nrm * val.y);
    unsafeAtomicAdd(od + v * 4 + 2, nrm * val.z);
    unsafeAtomicAdd(od + v * 4 + 3, nrm * val.w);
  }
}

// ---------------------------------------------------------------------------
// Host orchestration
// ---------------------------------------------------------------------------
extern "C" void kernel_launch(void* const* d_in, const int* in_sizes, int n_in,
                              void* d_out, int out_size, void* d_ws, size_t ws_size,
                              hipStream_t stream) {
  const float*     x  = (const float*)d_in[0];
  const long long* ei = (const long long*)d_in[1];
  const float*     ew = (const float*)d_in[2];
  const float*     W1 = (const float*)d_in[3];
  const float*     b1 = (const float*)d_in[4];
  const float*     W2 = (const float*)d_in[5];
  const float*     b2 = (const float*)d_in[6];
  float* out = (float*)d_out;

  // Workspace layout (bytes)
  char* ws = (char*)d_ws;
  float*          dinv = (float*)(ws + 0);                        // 400 KB
  float*          hbuf = (float*)(ws + (1u << 20));               // 102.4 MB (h1, then h2)
  float*          agg  = (float*)(ws + (1u << 20) + (100u << 20));// 102.4 MB (layer-1 pre-GELU)
  unsigned short* xpk  = (unsigned short*)(ws + (1u << 20) + (200u << 20)); // 51.2 MB packed bf16
  unsigned short* wpk  = (unsigned short*)(ws + (1u << 20) + (250u << 20)); // 128 KB packed bf16

  const int B = 256;

  // 1) symmetric normalization
  deg_init_k<<<(NN + B - 1) / B, B, 0, stream>>>(dinv);
  deg_scatter_k<<<(NE + B - 1) / B, B, 0, stream>>>(ei, ew, dinv);
  dinv_k<<<(NN + B - 1) / B, B, 0, stream>>>(dinv);

  // 2) layer 1: pack -> WMMA GEMM -> aggregate
  pack_b_k<<<(KDIM * 256) / B, B, 0, stream>>>(W1, wpk, 256);
  pack_a_k<<<(MT * 4096) / B, B, 0, stream>>>(x, xpk, 0);
  {
    dim3 g((MT + 7) / 8, 2);   // 2 n-chunks of 128 cols
    gemm_wmma_k<256><<<g, B, 0, stream>>>((const uint4*)xpk, (const uint4*)wpk, hbuf);
  }
  agg_init_k<256><<<(NN * 256) / B, B, 0, stream>>>(hbuf, dinv, b1, agg);
  edge_scatter_k<256><<<(NE * 32) / B, B, 0, stream>>>(ei, ew, dinv, hbuf, agg);

  // 3) layer 2: GELU fused into pack -> WMMA GEMM -> aggregate into d_out
  pack_a_k<<<(MT * 4096) / B, B, 0, stream>>>(agg, xpk, 1);
  pack_b_k<<<(KDIM * 128) / B, B, 0, stream>>>(W2, wpk, 128);
  {
    dim3 g((MT + 7) / 8, 1);   // 1 n-chunk of 128 cols
    gemm_wmma_k<128><<<g, B, 0, stream>>>((const uint4*)xpk, (const uint4*)wpk, hbuf);
  }
  agg_init_k<128><<<(NN * 128) / B, B, 0, stream>>>(hbuf, dinv, b2, out);
  edge_scatter_k<128><<<(NE * 32) / B, B, 0, stream>>>(ei, ew, dinv, hbuf, out);

  (void)in_sizes; (void)n_in; (void)out_size; (void)ws_size;
}
